// RelLearnbaleAttention_49091476194108
// MI455X (gfx1250) — matrix operand order repaired and checked by reading, not compiled, not run on hardware
//
#include <hip/hip_runtime.h>

typedef __attribute__((ext_vector_type(16))) __bf16 bf16x16;
typedef __attribute__((ext_vector_type(8)))  float  f32x8;
typedef unsigned short u16;

#define HD   16
#define DEP  64
#define BB   2
#define QL   1024
#define TL   2048
#define DM   1024

__device__ __forceinline__ u16 f2bf(float f) {
  unsigned int x = __float_as_uint(f);
  x += 0x7fffu + ((x >> 16) & 1u);
  return (u16)(x >> 16);
}

__device__ __forceinline__ bf16x16 frag16(uint4 lo, uint4 hi) {
  union U { unsigned int q[8]; bf16x16 v; __device__ U() {} } u;
  u.q[0] = lo.x; u.q[1] = lo.y; u.q[2] = lo.z; u.q[3] = lo.w;
  u.q[4] = hi.x; u.q[5] = hi.y; u.q[6] = hi.z; u.q[7] = hi.w;
  return u.v;
}

// ---------------- conversion / layout kernels -------------------
__global__ void cvt_strided(const float* __restrict__ src, u16* __restrict__ dst,
                            long long npb, long long sstride, long long dstride,
                            long long doff, int nb) {
  long long i = (long long)blockIdx.x * blockDim.x + threadIdx.x;
  if (i >= npb * nb) return;
  long long b = i / npb, r = i - b * npb;
  dst[b * dstride + doff + r] = f2bf(src[b * sstride + r]);
}

// qf: [B, L, H, D] f32 -> qu/qv: [B, H, L, D] bf16 with u/v head bias added
__global__ void prep_q(const float* __restrict__ qf, const float* __restrict__ u,
                       const float* __restrict__ v, u16* __restrict__ qu,
                       u16* __restrict__ qv) {
  long long i = (long long)blockIdx.x * blockDim.x + threadIdx.x;
  if (i >= (long long)BB * QL * HD * DEP) return;
  int d = (int)(i & 63), h = (int)((i >> 6) & 15);
  long long l = (i >> 10) & 1023, b = i >> 20;
  float x = qf[i];
  long long o = (((b * HD + h) * QL) + l) * DEP + d;
  qu[o] = f2bf(x + u[h * DEP + d]);
  qv[o] = f2bf(x + v[h * DEP + d]);
}

// kf/vf: [B, T, H, D] f32 -> [B, H, T, D] bf16
__global__ void prep_kv(const float* __restrict__ kf, const float* __restrict__ vf,
                        u16* __restrict__ kb, u16* __restrict__ vb) {
  long long i = (long long)blockIdx.x * blockDim.x + threadIdx.x;
  if (i >= (long long)BB * TL * HD * DEP) return;
  int d = (int)(i & 63), h = (int)((i >> 6) & 15);
  long long t = (i >> 10) & 2047, b = i >> 21;
  long long o = (((b * HD + h) * TL) + t) * DEP + d;
  kb[o] = f2bf(kf[i]);
  vb[o] = f2bf(vf[i]);
}

// pf: [T, H, D] f32 -> [H, T, D] bf16
__global__ void prep_p(const float* __restrict__ pf, u16* __restrict__ pb) {
  long long i = (long long)blockIdx.x * blockDim.x + threadIdx.x;
  if (i >= (long long)TL * HD * DEP) return;
  int d = (int)(i & 63), h = (int)((i >> 6) & 15);
  long long t = i >> 10;
  pb[((long long)h * TL + t) * DEP + d] = f2bf(pf[i]);
}

// ---------------- generic bf16 WMMA GEMM: C = A*B (+bias) -------------------
// 256 threads = 8 waves; 128x128 tile; k-step 32; each wave owns 32x64.
// Double-buffered LDS: global loads for tile i+1 are issued before the WMMAs
// of tile i, committed to the ping-pong buffer afterwards; one barrier/step.
__global__ __launch_bounds__(256) void gemm_bf16(
    const u16* __restrict__ A, const u16* __restrict__ Bm,
    const float* __restrict__ bias, float* __restrict__ C,
    int M, int N, int K, int lda, int ldb, int ldc,
    long long a_bs, long long b_bs, int c_zdiv, long long c_bs1, long long c_bs2) {
  int z = blockIdx.z;
  const u16* Ab = A + (long long)z * a_bs;
  const u16* Bb = Bm + (long long)z * b_bs;
  float* Cb = C + (long long)(z / c_zdiv) * c_bs1 + (long long)(z % c_zdiv) * c_bs2;
  int m0 = blockIdx.y * 128, n0 = blockIdx.x * 128;
  __shared__ __align__(16) u16 lA[2][128][40];   // [m][k], rows padded to 80B
  __shared__ __align__(16) u16 lB[2][128][40];   // transposed: [n][k]
  int tid = threadIdx.x, lane = tid & 31, w = tid >> 5;
  int wm = (w & 3) * 32, wn = (w >> 2) * 64;
  int ml = lane & 15, hs = lane >> 4;

  uint4 ra[2], rb[2];
  auto gload = [&](int k0) {
#pragma unroll
    for (int i = 0; i < 2; ++i) {
      int c = tid * 2 + i;
      int r = c >> 2, kc = (c & 3) * 8;          // A tile 128x32
      ra[i] = make_uint4(0u, 0u, 0u, 0u);
      if (m0 + r < M) ra[i] = *(const uint4*)(Ab + (long long)(m0 + r) * lda + k0 + kc);
      int k = c >> 4, n8 = (c & 15) * 8;         // B tile 32x128
      rb[i] = make_uint4(0u, 0u, 0u, 0u);
      if (n0 + n8 + 7 < N) rb[i] = *(const uint4*)(Bb + (long long)(k0 + k) * ldb + n0 + n8);
    }
  };
  auto lstore = [&](int bsel) {
#pragma unroll
    for (int i = 0; i < 2; ++i) {
      int c = tid * 2 + i;
      int r = c >> 2, kc = (c & 3) * 8;
      *(uint4*)&lA[bsel][r][kc] = ra[i];
      int k = c >> 4, n8 = (c & 15) * 8;
      const u16* pv = (const u16*)&rb[i];
#pragma unroll
      for (int j = 0; j < 8; ++j) lB[bsel][n8 + j][k] = pv[j];
    }
  };

  f32x8 acc[2][4] = {};
  gload(0);
  lstore(0);
  __syncthreads();
  int buf = 0;
  for (int k0 = 0; k0 < K; k0 += 32) {
    bool nxt = (k0 + 32) < K;
    if (nxt) gload(k0 + 32);                       // hide HBM latency behind WMMAs
    if (k0 + 64 < K) {                             // prime tile i+2 into caches
      int pr = m0 + (tid >> 1);
      if (pr < M) __builtin_prefetch(Ab + (long long)pr * lda + k0 + 64, 0, 1);
      int pk = k0 + 64 + (tid >> 4), pn = n0 + (tid & 15) * 8;
      if (pk < K && pn + 7 < N) __builtin_prefetch(Bb + (long long)pk * ldb + pn, 0, 1);
    }
    bf16x16 af[2], bfv[4];
#pragma unroll
    for (int f = 0; f < 2; ++f) {   // A frags: lane row = ml, k at hs*8 / 16+hs*8
      const u16* p = &lA[buf][wm + f * 16 + ml][0];
      af[f] = frag16(*(const uint4*)(p + hs * 8), *(const uint4*)(p + 16 + hs * 8));
    }
#pragma unroll
    for (int g = 0; g < 4; ++g) {   // B frags: lane col = ml, k at hs*8 / 16+hs*8
      const u16* p = &lB[buf][wn + g * 16 + ml][0];
      bfv[g] = frag16(*(const uint4*)(p + hs * 8), *(const uint4*)(p + 16 + hs * 8));
    }
#pragma unroll
    for (int f = 0; f < 2; ++f)
#pragma unroll
      for (int g = 0; g < 4; ++g)
        acc[f][g] = __builtin_amdgcn_wmma_f32_16x16x32_bf16(
            false, af[f], false, bfv[g], (short)0, acc[f][g], false, false);
    if (nxt) lstore(buf ^ 1);
    __syncthreads();
    buf ^= 1;
  }
#pragma unroll
  for (int f = 0; f < 2; ++f) {
    int rbs = m0 + wm + f * 16 + hs * 8;
#pragma unroll
    for (int g = 0; g < 4; ++g) {
      int col = n0 + wn + g * 16 + ml;
      if (col < N) {
        float bvv = bias ? bias[col] : 0.0f;
#pragma unroll
        for (int r = 0; r < 8; ++r) {
          int row = rbs + r;
          if (row < M) Cb[(long long)row * ldc + col] = acc[f][g][r] + bvv;
        }
      }
    }
  }
}

// ---------------- fused content+positional logits -------------------
// grid (T/64, L/64, B*H), 128 threads = 4 waves; wave w owns q rows [l0+16w, +16).
// content = (q+u)@K^T via WMMA; pos = band GEMM (q+v)@P^T over 128 j-rows,
// then the Transformer-XL skew becomes an always-in-range LDS gather:
//   pos_att[l,t] = rel[l, t+1023-l]  (rows j>=T zero-filled, matching the
//   zero column; wrap-region entries are semantically masked and set to 0).
__global__ __launch_bounds__(128) void attn_logits(
    const u16* __restrict__ qu, const u16* __restrict__ qv,
    const u16* __restrict__ kk, const u16* __restrict__ pp,
    const int* __restrict__ mask, float* __restrict__ wts) {
  int z = blockIdx.z, h = z & 15;
  int t0 = blockIdx.x * 64, l0 = blockIdx.y * 64;
  const u16* quz = qu + (long long)z * QL * DEP;
  const u16* qvz = qv + (long long)z * QL * DEP;
  const u16* kz  = kk + (long long)z * TL * DEP;
  const u16* pz  = pp + (long long)h * TL * DEP;
  int tid = threadIdx.x, lane = tid & 31, w = tid >> 5;
  int ml = lane & 15, hs = lane >> 4;
  int rowb = l0 + w * 16;

  bf16x16 aqu[2], aqv[2];
#pragma unroll
  for (int s = 0; s < 2; ++s) {
    const u16* pa = quz + (long long)(rowb + ml) * DEP + 32 * s + hs * 8;
    aqu[s] = frag16(*(const uint4*)pa, *(const uint4*)(pa + 16));
    const u16* pb = qvz + (long long)(rowb + ml) * DEP + 32 * s + hs * 8;
    aqv[s] = frag16(*(const uint4*)pb, *(const uint4*)(pb + 16));
  }

  f32x8 S[4] = {};
#pragma unroll
  for (int f = 0; f < 4; ++f)
#pragma unroll
    for (int s = 0; s < 2; ++s) {
      const u16* pk = kz + (long long)(t0 + f * 16 + ml) * DEP + 32 * s + hs * 8;
      bf16x16 b = frag16(*(const uint4*)pk, *(const uint4*)(pk + 16));
      S[f] = __builtin_amdgcn_wmma_f32_16x16x32_bf16(
          false, aqu[s], false, b, (short)0, S[f], false, false);
    }

  int jb = t0 - l0 + (QL - 64);   // band base: always >= 0
  f32x8 R[8] = {};
#pragma unroll
  for (int g = 0; g < 8; ++g) {
    int j = jb + g * 16 + ml;
#pragma unroll
    for (int s = 0; s < 2; ++s) {
      uint4 lo = make_uint4(0u, 0u, 0u, 0u), hi = lo;
      if (j < TL) {
        const u16* pp2 = pz + (long long)j * DEP + 32 * s + hs * 8;
        lo = *(const uint4*)pp2; hi = *(const uint4*)(pp2 + 16);
      }
      bf16x16 b = frag16(lo, hi);
      R[g] = __builtin_amdgcn_wmma_f32_16x16x32_bf16(
          false, aqv[s], false, b, (short)0, R[g], false, false);
    }
  }

  __shared__ float lR[4][16][128];
#pragma unroll
  for (int g = 0; g < 8; ++g)
#pragma unroll
    for (int r = 0; r < 8; ++r)
      lR[w][r + 8 * hs][g * 16 + ml] = R[g][r];
  __syncthreads();

#pragma unroll
  for (int f = 0; f < 4; ++f) {
    int gt = t0 + f * 16 + ml;
#pragma unroll
    for (int r = 0; r < 8; ++r) {
      int m = r + 8 * hs;
      int col = f * 16 + ml + 63 - 16 * w - m;   // in [0,127] by construction
      float val = (S[f][r] + lR[w][m][col]) * 0.125f;  // 1/sqrt(64)
      int gl = rowb + m;
      if (mask[(long long)gl * TL + gt] == 0) val = -1e9f;
      wts[((long long)z * QL + gl) * TL + gt] = val;
    }
  }
}

// ---------------- row softmax in place, one wave per 2048-wide row ----------
__global__ __launch_bounds__(256) void softmax_rows(float* __restrict__ wts) {
  int w = threadIdx.x >> 5, lane = threadIdx.x & 31;
  long long row = (long long)blockIdx.x * 8 + w;
  float* p = wts + row * TL;
  float v[64];
  float mx = -3.0e38f;
#pragma unroll
  for (int i = 0; i < 64; ++i) { v[i] = p[lane + 32 * i]; mx = fmaxf(mx, v[i]); }
#pragma unroll
  for (int o = 16; o > 0; o >>= 1) mx = fmaxf(mx, __shfl_xor(mx, o, 32));
  float s = 0.f;
#pragma unroll
  for (int i = 0; i < 64; ++i) { v[i] = __expf(v[i] - mx); s += v[i]; }
#pragma unroll
  for (int o = 16; o > 0; o >>= 1) s += __shfl_xor(s, o, 32);
  float inv = 1.0f / s;
#pragma unroll
  for (int i = 0; i < 64; ++i) p[lane + 32 * i] = v[i] * inv;
}

// ---------------- host orchestration -------------------
extern "C" void kernel_launch(void* const* d_in, const int* in_sizes, int n_in,
                              void* d_out, int out_size, void* d_ws, size_t ws_size,
                              hipStream_t stream) {
  const float* query = (const float*)d_in[0];
  const float* key   = (const float*)d_in[1];
  const float* value = (const float*)d_in[2];
  const float* pos   = (const float*)d_in[3];
  const float* u     = (const float*)d_in[4];
  const float* vhv   = (const float*)d_in[5];
  const float* mem   = (const float*)d_in[6];
  const int*   mask  = (const int*)d_in[7];
  const float* Wq = (const float*)d_in[8];  const float* bq = (const float*)d_in[9];
  const float* Wk = (const float*)d_in[10]; const float* bk = (const float*)d_in[11];
  const float* Wv = (const float*)d_in[12]; const float* bv = (const float*)d_in[13];
  const float* Wp = (const float*)d_in[14]; const float* bp = (const float*)d_in[15];
  const float* Wo = (const float*)d_in[16]; const float* bo = (const float*)d_in[17];

  float* out = (float*)d_out;                        // [B, L, DM]
  float* wts = out + (long long)BB * QL * DM;        // [B, H, L, T]

  char* p = (char*)d_ws;
  auto carve = [&](size_t bytes) -> char* {
    char* r = p; p += (bytes + 255) & ~(size_t)255; return r;
  };
  u16* qB   = (u16*)carve((size_t)BB * QL * DM * 2);
  u16* kinB = (u16*)carve((size_t)BB * TL * DM * 2);
  u16* vinB = (u16*)carve((size_t)BB * TL * DM * 2);
  u16* posB = (u16*)carve((size_t)TL * DM * 2);
  u16* WqB  = (u16*)carve((size_t)DM * DM * 2);
  u16* WkB  = (u16*)carve((size_t)DM * DM * 2);
  u16* WvB  = (u16*)carve((size_t)DM * DM * 2);
  u16* WpB  = (u16*)carve((size_t)DM * DM * 2);
  u16* WoB  = (u16*)carve((size_t)DM * DM * 2);
  float* qf = (float*)carve((size_t)BB * QL * DM * 4);
  float* kf = (float*)carve((size_t)BB * TL * DM * 4);
  float* vf = (float*)carve((size_t)BB * TL * DM * 4);
  float* pf = (float*)carve((size_t)TL * DM * 4);
  u16* quB  = (u16*)carve((size_t)BB * HD * QL * DEP * 2);
  u16* qvB  = (u16*)carve((size_t)BB * HD * QL * DEP * 2);
  u16* kbh  = (u16*)carve((size_t)BB * HD * TL * DEP * 2);
  u16* vbh  = (u16*)carve((size_t)BB * HD * TL * DEP * 2);
  u16* pbh  = (u16*)carve((size_t)HD * TL * DEP * 2);
  u16* wB   = (u16*)carve((size_t)BB * HD * QL * TL * 2);
  float* attnf = (float*)carve((size_t)BB * QL * DM * 4);
  u16* attnB   = (u16*)carve((size_t)BB * QL * DM * 2);

  const int TPB = 256;
  auto blks = [](long long n) { return (unsigned)((n + 255) / 256); };
  long long nQ = (long long)BB * QL * DM;   // 2M
  long long nT = (long long)BB * TL * DM;   // 4M
  long long nP = (long long)TL * DM;        // 2M
  long long nW = (long long)DM * DM;        // 1M
  long long half = (long long)QL * DM;      // per-batch rows of mem/key

  // fp32 -> bf16 (and mem||key / mem||value concat)
  cvt_strided<<<blks(nQ), TPB, 0, stream>>>(query, qB, nQ, 0, 0, 0, 1);
  cvt_strided<<<blks(nQ), TPB, 0, stream>>>(mem,   kinB, half, half, (long long)TL * DM, 0,    BB);
  cvt_strided<<<blks(nQ), TPB, 0, stream>>>(key,   kinB, half, half, (long long)TL * DM, half, BB);
  cvt_strided<<<blks(nQ), TPB, 0, stream>>>(mem,   vinB, half, half, (long long)TL * DM, 0,    BB);
  cvt_strided<<<blks(nQ), TPB, 0, stream>>>(value, vinB, half, half, (long long)TL * DM, half, BB);
  cvt_strided<<<blks(nP), TPB, 0, stream>>>(pos, posB, nP, 0, 0, 0, 1);
  cvt_strided<<<blks(nW), TPB, 0, stream>>>(Wq, WqB, nW, 0, 0, 0, 1);
  cvt_strided<<<blks(nW), TPB, 0, stream>>>(Wk, WkB, nW, 0, 0, 0, 1);
  cvt_strided<<<blks(nW), TPB, 0, stream>>>(Wv, WvB, nW, 0, 0, 0, 1);
  cvt_strided<<<blks(nW), TPB, 0, stream>>>(Wp, WpB, nW, 0, 0, 0, 1);
  cvt_strided<<<blks(nW), TPB, 0, stream>>>(Wo, WoB, nW, 0, 0, 0, 1);

  // projections
  gemm_bf16<<<dim3(DM / 128, BB * QL / 128, 1), 256, 0, stream>>>(
      qB, WqB, bq, qf, BB * QL, DM, DM, DM, DM, DM, 0, 0, 1, 0, 0);
  gemm_bf16<<<dim3(DM / 128, BB * TL / 128, 1), 256, 0, stream>>>(
      kinB, WkB, bk, kf, BB * TL, DM, DM, DM, DM, DM, 0, 0, 1, 0, 0);
  gemm_bf16<<<dim3(DM / 128, BB * TL / 128, 1), 256, 0, stream>>>(
      vinB, WvB, bv, vf, BB * TL, DM, DM, DM, DM, DM, 0, 0, 1, 0, 0);
  gemm_bf16<<<dim3(DM / 128, TL / 128, 1), 256, 0, stream>>>(
      posB, WpB, bp, pf, TL, DM, DM, DM, DM, DM, 0, 0, 1, 0, 0);

  // head-major bf16 operands
  prep_q<<<blks(nQ), TPB, 0, stream>>>(qf, u, vhv, quB, qvB);
  prep_kv<<<blks(nT), TPB, 0, stream>>>(kf, vf, kbh, vbh);
  prep_p<<<blks(nP), TPB, 0, stream>>>(pf, pbh);

  // logits (content + skewed pos) straight into d_out weights region
  attn_logits<<<dim3(TL / 64, QL / 64, BB * HD), 128, 0, stream>>>(
      quB, qvB, kbh, pbh, mask, wts);

  // softmax in place
  softmax_rows<<<(BB * HD * QL) / 8, 256, 0, stream>>>(wts);

  // weights @ V  (batched over B*H, N=64 guarded)
  long long nwts = (long long)BB * HD * QL * TL;
  cvt_strided<<<blks(nwts), TPB, 0, stream>>>(wts, wB, nwts, 0, 0, 0, 1);
  gemm_bf16<<<dim3(1, QL / 128, BB * HD), 256, 0, stream>>>(
      wB, vbh, nullptr, attnf, QL, DEP, TL, TL, DEP, DM,
      (long long)QL * TL, (long long)TL * DEP, HD, (long long)QL * DM, (long long)DEP);

  // output projection
  cvt_strided<<<blks(nQ), TPB, 0, stream>>>(attnf, attnB, nQ, 0, 0, 0, 1);
  gemm_bf16<<<dim3(DM / 128, BB * QL / 128, 1), 256, 0, stream>>>(
      attnB, WoB, bo, out, BB * QL, DM, DM, DM, DM, DM, 0, 0, 1, 0, 0);
}
